// Dynamic_conv2d_17806934409821
// MI455X (gfx1250) — matrix operand order repaired and checked by reading, not compiled, not run on hardware
//
#include <hip/hip_runtime.h>
#include <hip/hip_bf16.h>

#define C_INC 128
#define C_OUTC 256
#define HGT 56
#define WID 56
#define BATCH 32
#define KEXP 8
#define HP 58                  // padded spatial edge
#define HPP (HP * HP)          // 3364 = padded plane
#define KDIM (C_INC * 9)       // 1152 GEMM reduction dim, ordered kk = rs*128 + c
#define NPIX (HGT * WID)       // 3136

typedef __attribute__((ext_vector_type(16))) __bf16 v16bf;
typedef __attribute__((ext_vector_type(8)))  float  v8f;
typedef __attribute__((ext_vector_type(4)))  unsigned int uint4v;
typedef __attribute__((ext_vector_type(4)))  int          v4i;

// gfx1250 async LDS-DMA path (ASYNCcnt); signature probe-confirmed:
// (int4 AS1* src, int4 AS3* dst, imm offset, imm cpol)
#if __has_builtin(__builtin_amdgcn_global_load_async_to_lds_b128)
#define HAS_ASYNC_LDS 1
typedef __attribute__((address_space(1))) v4i* gv4i_p;
typedef __attribute__((address_space(3))) v4i* lv4i_p;
#else
#define HAS_ASYNC_LDS 0
#endif

#if __has_builtin(__builtin_amdgcn_s_wait_asynccnt)
#define ASYNC_WAIT0() __builtin_amdgcn_s_wait_asynccnt(0)
#else
#define ASYNC_WAIT0() asm volatile("s_wait_asynccnt 0" ::: "memory")
#endif

__device__ __forceinline__ unsigned short f2bf(float f) {
  unsigned int u = __float_as_uint(f);
  u += 0x7FFFu + ((u >> 16) & 1u);   // round-to-nearest-even
  return (unsigned short)(u >> 16);
}

// ---------------------------------------------------------------------------
// K1: GAP per (b,c) plane + zero-padded bf16 copy of x.  grid(C_in,B)x256
// ---------------------------------------------------------------------------
__global__ __launch_bounds__(256) void gap_pad_kernel(
    const float* __restrict__ x, float* __restrict__ gap,
    unsigned short* __restrict__ xbf) {
  const int c = blockIdx.x, b = blockIdx.y, t = threadIdx.x;
  const float* src = x + (size_t)(b * C_INC + c) * NPIX;
  unsigned short* dst = xbf + (size_t)(b * C_INC + c) * HPP;

  float s = 0.f;
  for (int i = t; i < NPIX; i += 256) s += src[i];

  for (int i = t; i < HPP; i += 256) {
    const int yy = i / HP, xx = i - yy * HP;
    unsigned short v = 0;
    if (yy >= 1 && yy <= HGT && xx >= 1 && xx <= WID)
      v = f2bf(src[(yy - 1) * WID + (xx - 1)]);
    dst[i] = v;
  }

  __shared__ float red[256];
  red[t] = s;
  __syncthreads();
  for (int o = 128; o > 0; o >>= 1) {
    if (t < o) red[t] += red[t + o];
    __syncthreads();
  }
  if (t == 0) gap[b * C_INC + c] = red[0] * (1.0f / NPIX);
}

// ---------------------------------------------------------------------------
// K2: attn[b,k] = sigmoid(gap[b,:]·fc_w[k,:] + fc_b[k]).  1 block x 256 = B*K
// ---------------------------------------------------------------------------
__global__ __launch_bounds__(256) void attn_kernel(
    const float* __restrict__ gap, const float* __restrict__ fc_w,
    const float* __restrict__ fc_b, float* __restrict__ attn) {
  const int t = threadIdx.x, b = t >> 3, k = t & 7;
  const float* g = gap + b * C_INC;
  const float* w = fc_w + k * C_INC;
  float s = fc_b[k];
  #pragma unroll 8
  for (int c = 0; c < C_INC; ++c) s += g[c] * w[c];
  attn[t] = 1.0f / (1.0f + __expf(-s));
}

// ---------------------------------------------------------------------------
// K3: aggb[b,o] = sum_k attn[b,k]*bias[k,o].  grid(B) x 256
// ---------------------------------------------------------------------------
__global__ __launch_bounds__(256) void aggb_kernel(
    const float* __restrict__ attn, const float* __restrict__ bias,
    float* __restrict__ aggb) {
  const int b = blockIdx.x, o = threadIdx.x;
  float s = 0.f;
  #pragma unroll
  for (int k = 0; k < KEXP; ++k) s += attn[b * KEXP + k] * bias[k * C_OUTC + o];
  aggb[b * C_OUTC + o] = s;
}

// ---------------------------------------------------------------------------
// K4: aggregated weights -> bf16, PERMUTED layout A[b][o][kk], kk = rs*128 + c
// grid (288, B) x 256, 4 consecutive outputs per thread
// ---------------------------------------------------------------------------
__global__ __launch_bounds__(256) void aggw_kernel(
    const float* __restrict__ attn, const float* __restrict__ weight,
    unsigned short* __restrict__ aggw) {
  const int b = blockIdx.y;
  __shared__ float a[KEXP];
  if (threadIdx.x < KEXP) a[threadIdx.x] = attn[b * KEXP + threadIdx.x];
  __syncthreads();

  const size_t EW = (size_t)C_OUTC * KDIM;          // 294912 per expert
  const int idx = (blockIdx.x * 256 + threadIdx.x) * 4;  // output kk-run of 4
  const int o   = idx / KDIM;
  const int rem = idx - o * KDIM;
  const int rs  = rem >> 7;          // tap index 0..8
  const int c0  = rem & 127;         // channel base (multiple of 4)

  const float* wb = weight + o * KDIM + c0 * 9 + rs;   // input: o*1152 + c*9 + rs
  float a0 = 0.f, a1 = 0.f, a2 = 0.f, a3 = 0.f;
  #pragma unroll
  for (int k = 0; k < KEXP; ++k) {
    const float ak = a[k];
    const float* wk = wb + k * EW;
    a0 += ak * wk[0];  a1 += ak * wk[9];
    a2 += ak * wk[18]; a3 += ak * wk[27];
  }
  const unsigned int lo = (unsigned int)f2bf(a0) | ((unsigned int)f2bf(a1) << 16);
  const unsigned int hi = (unsigned int)f2bf(a2) | ((unsigned int)f2bf(a3) << 16);
  *reinterpret_cast<uint2*>(aggw + b * EW + idx) = make_uint2(lo, hi);
}

// ---------------------------------------------------------------------------
// K5: implicit-GEMM conv, v_wmma_f32_16x16x32_bf16, double-buffered LDS,
// async LDS-DMA for the A tile.
//   block tile 256x64 (full C_out), KC=32; 8 waves as 4(M) x 2(N);
//   wave tile 64x32 = 4x2 WMMA -> 12 ds_loads per 8 WMMAs.
// grid (49, 1, B)
// ---------------------------------------------------------------------------
#define MT 256
#define NT 64
#define KC 32
#define APAD 40          // LDS row pitch in bf16 (bank-conflict pad)
#define NKT (KDIM / KC)  // 36

__global__ __launch_bounds__(256) void dynconv_gemm(
    const unsigned short* __restrict__ aggw,
    const unsigned short* __restrict__ xbf,
    const float* __restrict__ aggb,
    float* __restrict__ out) {
  const int b  = blockIdx.z;
  const int n0 = blockIdx.x * NT;
  const int t  = threadIdx.x;
  const int lane = t & 31, wave = t >> 5;
  const int wm = wave & 3, wn = wave >> 2;   // 4x2 waves; wave tile 64x32

  __shared__ unsigned short As[2][MT * APAD];   // 2 x 20480 B
  __shared__ unsigned short Bs[2][NT * APAD];   // 2 x  5120 B

  const unsigned short* gA = aggw + (size_t)b * (C_OUTC * KDIM);
  const unsigned short* gX = xbf + (size_t)b * (C_INC * HPP);

  // A staging: 1024 x 16B chunks, 4 per thread (rows arow0 + {0,64,128,192})
  const int arow0 = t >> 2, acol = (t & 3) * 8;

  // B staging: thread -> (n, kk-octet); addr = bbase + j*HPP + uoff(kt)
  const int bn  = t & 63;
  const int bks = (t >> 6) * 8;
  const int p   = n0 + bn;
  const int py  = p / WID, px = p - py * WID;
  const int bbase = bks * HPP + py * HP + px;   // thread-invariant

  auto loadA = [&](int kt, int buf) {
    const int kc = kt * KC;
    #pragma unroll
    for (int i = 0; i < 4; ++i) {
      const int row = arow0 + i * 64;
#if HAS_ASYNC_LDS
      v4i* s = (v4i*)(gA + (size_t)row * KDIM + kc + acol);
      v4i* d = (v4i*)&As[buf][row * APAD + acol];
      __builtin_amdgcn_global_load_async_to_lds_b128((gv4i_p)s, (lv4i_p)d, 0, 0);
#else
      const uint4v v = *reinterpret_cast<const uint4v*>(
          gA + (size_t)row * KDIM + kc + acol);
      *reinterpret_cast<uint4v*>(&As[buf][row * APAD + acol]) = v;
#endif
    }
  };

  auto loadB = [&](int kt, int buf) {
    const int rs = kt >> 2;                 // tap 0..8 (uniform)
    const int r  = rs / 3, s2 = rs - 3 * r; // uniform scalar math
    const int c0 = (kt & 3) << 5;           // channel base of chunk
    const int uoff = c0 * HPP + r * HP + s2;
    uint4v w;
    #pragma unroll
    for (int jj = 0; jj < 4; ++jj) {        // pack pairs -> d16/d16_hi loads
      const unsigned int lo = gX[bbase + (2 * jj) * HPP + uoff];
      const unsigned int hi = gX[bbase + (2 * jj + 1) * HPP + uoff];
      w[jj] = lo | (hi << 16);
    }
    *reinterpret_cast<uint4v*>(&Bs[buf][bn * APAD + bks]) = w;
  };

  v8f acc[4][2] = {};

  loadA(0, 0);
  loadB(0, 0);

  for (int kt = 0; kt < NKT; ++kt) {
    const int buf = kt & 1;
#if HAS_ASYNC_LDS
    ASYNC_WAIT0();            // drain this wave's async DMAs before the barrier
#endif
    __syncthreads();          // tile kt visible to all waves

    if (kt + 1 < NKT) {       // overlap next-tile staging with WMMA
      loadA(kt + 1, buf ^ 1);
      loadB(kt + 1, buf ^ 1);
    }

    // per-lane fragments (documented 16-bit A/B WMMA VGPR layouts)
    const int khalf = lane >> 4;
    const int lr    = lane & 15;
    union Frag { uint4v u[2]; v16bf v; };
    Frag af[4], bfv[2];
    #pragma unroll
    for (int i = 0; i < 4; ++i) {
      const int row = wm * 64 + i * 16 + lr;
      af[i].u[0] = *reinterpret_cast<const uint4v*>(&As[buf][row * APAD + khalf * 8]);
      af[i].u[1] = *reinterpret_cast<const uint4v*>(&As[buf][row * APAD + 16 + khalf * 8]);
    }
    #pragma unroll
    for (int j = 0; j < 2; ++j) {
      const int col = wn * 32 + j * 16 + lr;
      bfv[j].u[0] = *reinterpret_cast<const uint4v*>(&Bs[buf][col * APAD + khalf * 8]);
      bfv[j].u[1] = *reinterpret_cast<const uint4v*>(&Bs[buf][col * APAD + 16 + khalf * 8]);
    }

    #pragma unroll
    for (int i = 0; i < 4; ++i)
      #pragma unroll
      for (int j = 0; j < 2; ++j)
        acc[i][j] = __builtin_amdgcn_wmma_f32_16x16x32_bf16(
            false, af[i].v, false, bfv[j].v, (short)0, acc[i][j], false, false);

    __syncthreads();          // all reads of buf done before it is re-staged
  }

  // epilogue: C/D layout -> (m = v + (lane>=16)*8, n = lane&15), + bias
  const int nn   = lane & 15;
  const int moff = (lane >> 4) * 8;
  #pragma unroll
  for (int i = 0; i < 4; ++i) {
    #pragma unroll
    for (int j = 0; j < 2; ++j) {
      const int pp = n0 + wn * 32 + j * 16 + nn;
      #pragma unroll
      for (int v = 0; v < 8; ++v) {
        const int m = wm * 64 + i * 16 + moff + v;
        out[((size_t)(b * C_OUTC + m)) * NPIX + pp] =
            acc[i][j][v] + aggb[b * C_OUTC + m];
      }
    }
  }
}

// ---------------------------------------------------------------------------
extern "C" void kernel_launch(void* const* d_in, const int* in_sizes, int n_in,
                              void* d_out, int out_size, void* d_ws, size_t ws_size,
                              hipStream_t stream) {
  const float* x      = (const float*)d_in[0];
  const float* fc_w   = (const float*)d_in[1];
  const float* fc_b   = (const float*)d_in[2];
  const float* weight = (const float*)d_in[3];
  const float* bias   = (const float*)d_in[4];
  float* out = (float*)d_out;

  char* ws = (char*)d_ws;
  // ws layout: gap f32 @0 (16KB) | attn f32 @16384 (1KB) | aggb f32 @17408 (32KB)
  //            xbf bf16 @50176 (27.56MB) | aggw bf16 @27608064 (18.87MB)
  float* gap  = (float*)(ws + 0);
  float* attn = (float*)(ws + 16384);
  float* aggb = (float*)(ws + 17408);
  unsigned short* xbf  = (unsigned short*)(ws + 50176);
  unsigned short* aggw = (unsigned short*)(ws + 27608064ull);

  gap_pad_kernel<<<dim3(C_INC, BATCH), 256, 0, stream>>>(x, gap, xbf);
  attn_kernel<<<1, 256, 0, stream>>>(gap, fc_w, fc_b, attn);
  aggb_kernel<<<BATCH, 256, 0, stream>>>(attn, bias, aggb);
  aggw_kernel<<<dim3(288, BATCH), 256, 0, stream>>>(attn, weight, aggw);
  dynconv_gemm<<<dim3(NPIX / NT, 1, BATCH), 256, 0, stream>>>(
      aggw, xbf, aggb, out);
}
